// LinearAttention_55722905699326
// MI455X (gfx1250) — matrix operand order repaired and checked by reading, not compiled, not run on hardware
//
#include <hip/hip_runtime.h>

typedef __bf16 bf16;
typedef __attribute__((ext_vector_type(16))) __bf16 v16bf;
typedef __attribute__((ext_vector_type(8)))  __bf16 v8bf;
typedef __attribute__((ext_vector_type(8)))  float  v8f;
typedef int v4i __attribute__((vector_size(16)));

constexpr int Bc = 4, Sc = 4096, Hc = 1024, NHc = 16, HDc = 64;
constexpr float EPSc = 1e-8f;

#define WMMA_BF16(a, b, c) \
  __builtin_amdgcn_wmma_f32_16x16x32_bf16(false, (a), false, (b), (short)0, (c), false, false)

#define AS1 __attribute__((address_space(1)))
#define AS3 __attribute__((address_space(3)))

#if defined(__HIP_DEVICE_COMPILE__) && __has_builtin(__builtin_amdgcn_global_load_async_to_lds_b128)
#define HAVE_ASYNC 1
#else
#define HAVE_ASYNC 0
#endif

// 16-byte global -> LDS copy: async DMA (ASYNCcnt) when available.
__device__ __forceinline__ void copy16(const bf16* g, bf16* l) {
#if HAVE_ASYNC
  __builtin_amdgcn_global_load_async_to_lds_b128(
      (AS1 v4i*)(unsigned long long)(const void*)g,
      (AS3 v4i*)(unsigned)(unsigned long long)(void*)l, 0, 0);
#else
  *(v8bf*)l = *(const v8bf*)g;
#endif
}

__device__ __forceinline__ void wait_async() {
#if HAVE_ASYNC
#if __has_builtin(__builtin_amdgcn_s_wait_asynccnt)
  __builtin_amdgcn_s_wait_asynccnt(0);
#else
  asm volatile("s_wait_asynccnt 0" ::: "memory");
#endif
#endif
}

union FragU { v16bf v; uint4 q[2]; };

// A-matrix 16x32 bf16 fragment (ISA 7.12.2: lane L row M=L%16; half=L/16;
// elems 0..7 -> K = half*8+0..7, elems 8..15 -> K = 16+half*8+0..7)
__device__ __forceinline__ v16bf frag_a(const bf16* buf, int stride, int row0, int k0) {
  const int lane = threadIdx.x & 31;
  const bf16* p = buf + (row0 + (lane & 15)) * stride + k0 + ((lane >> 4) << 3);
  FragU f;
  f.q[0] = *(const uint4*)p;
  f.q[1] = *(const uint4*)(p + 16);
  return f.v;
}

// B-matrix 32x16 bf16 fragment: lane L column N=L%16; half=L/16;
// elems 0..7 -> K = half*8+0..7, elems 8..15 -> K = 16+half*8+0..7. buf stored N-major (N x K).
__device__ __forceinline__ v16bf frag_b(const bf16* buf, int stride, int col0, int k0) {
  const int lane = threadIdx.x & 31;
  const bf16* p = buf + (col0 + (lane & 15)) * stride + k0 + ((lane >> 4) << 3);
  FragU f;
  f.q[0] = *(const uint4*)p;
  f.q[1] = *(const uint4*)(p + 16);
  return f.v;
}

__device__ __forceinline__ void zero_acc(v8f& a) {
#pragma unroll
  for (int i = 0; i < 8; ++i) a[i] = 0.0f;
}

__device__ __forceinline__ v8bf pack8(float4 a, float4 b) {
  v8bf o;
  o[0] = (bf16)a.x; o[1] = (bf16)a.y; o[2] = (bf16)a.z; o[3] = (bf16)a.w;
  o[4] = (bf16)b.x; o[5] = (bf16)b.y; o[6] = (bf16)b.z; o[7] = (bf16)b.w;
  return o;
}

// cooperative bf16 global -> LDS tile copy, fully unrolled (async when available)
template <int ROWS, int COLS, int LDD>
__device__ __forceinline__ void stage_async(const bf16* __restrict__ src, int ldsrc,
                                            bf16* __restrict__ dst) {
  constexpr int CU = COLS / 8;
  static_assert((ROWS * CU) % 256 == 0, "tile not divisible");
#pragma unroll
  for (int it = 0; it < ROWS * CU / 256; ++it) {
    const int u = threadIdx.x + it * 256;
    const int r = u / CU;
    const int c = (u % CU) << 3;
    copy16(src + (size_t)r * ldsrc + c, dst + r * LDD + c);
  }
}

// transpose stage: src [64 x 64] fp32 row-major -> dst[col][row] bf16
__device__ __forceinline__ void stage_T64_f32(const float* src, int ldsrc, bf16* dst, int lddst) {
  const int r = threadIdx.x & 63;
  const int c0 = (threadIdx.x >> 6) << 4;
#pragma unroll
  for (int i = 0; i < 4; ++i) {
    float4 f = *(const float4*)(src + (size_t)r * ldsrc + c0 + i * 4);
    const int c = c0 + i * 4;
    dst[(c + 0) * lddst + r] = (bf16)f.x;
    dst[(c + 1) * lddst + r] = (bf16)f.y;
    dst[(c + 2) * lddst + r] = (bf16)f.z;
    dst[(c + 3) * lddst + r] = (bf16)f.w;
  }
}

// transpose stage: src [64 x 64] bf16 row-major -> dst[col][row] bf16
__device__ __forceinline__ void stage_T64_h(const bf16* src, int ldsrc, bf16* dst, int lddst) {
  const int r = threadIdx.x & 63;
  const int c0 = (threadIdx.x >> 6) << 4;
  v8bf x0 = *(const v8bf*)(src + (size_t)r * ldsrc + c0);
  v8bf x1 = *(const v8bf*)(src + (size_t)r * ldsrc + c0 + 8);
#pragma unroll
  for (int i = 0; i < 8; ++i) dst[(c0 + i) * lddst + r] = x0[i];
#pragma unroll
  for (int i = 0; i < 8; ++i) dst[(c0 + 8 + i) * lddst + r] = x1[i];
}

// ------- Y[M,N] = X[M,K] @ W[N,K]^T, double-buffered, async-staged -----------
template <bool A_F32, bool OUT_F32>
__global__ __launch_bounds__(256) void la_gemm(const void* __restrict__ Xv,
                                               const bf16* __restrict__ W,
                                               void* __restrict__ Yv,
                                               int M, int N, int K) {
  constexpr int TM = 128, TN = 128, TK = 32, LDT = TK + 8;
  __shared__ bf16 As[2][TM * LDT];
  __shared__ bf16 Bs[2][TN * LDT];
  const float* Xf = (const float*)Xv;
  const bf16* Xh = (const bf16*)Xv;
  const int bm = blockIdx.y * TM, bn = blockIdx.x * TN;
  const int tid = threadIdx.x, lane = tid & 31, wave = tid >> 5;
  const int wm = (wave & 1) * 64, wn = (wave >> 1) * 32;
  // per-thread staging coordinates: two 16B units (rows r0, r0+64)
  const int r0 = tid >> 2, c0 = (tid & 3) << 3;
  const int r1 = r0 + 64;

  v8f acc[4][2];
#pragma unroll
  for (int i = 0; i < 4; ++i) { zero_acc(acc[i][0]); zero_acc(acc[i][1]); }

  // prologue: stage k0 = 0 into buffer 0
  if constexpr (A_F32) {
    const float4* s0p = (const float4*)(Xf + (size_t)(bm + r0) * K + c0);
    const float4* s1p = (const float4*)(Xf + (size_t)(bm + r1) * K + c0);
    *(v8bf*)(As[0] + r0 * LDT + c0) = pack8(s0p[0], s0p[1]);
    *(v8bf*)(As[0] + r1 * LDT + c0) = pack8(s1p[0], s1p[1]);
  } else {
    copy16(Xh + (size_t)(bm + r0) * K + c0, As[0] + r0 * LDT + c0);
    copy16(Xh + (size_t)(bm + r1) * K + c0, As[0] + r1 * LDT + c0);
  }
  copy16(W + (size_t)(bn + r0) * K + c0, Bs[0] + r0 * LDT + c0);
  copy16(W + (size_t)(bn + r1) * K + c0, Bs[0] + r1 * LDT + c0);

  int cur = 0;
  for (int k0 = 0; k0 < K; k0 += TK) {
    wait_async();
    __syncthreads();
    const bool nxt = (k0 + TK) < K;
    bf16* an = As[cur ^ 1];
    bf16* bnx = Bs[cur ^ 1];
    float4 p00, p01, p10, p11;
    if (nxt) {
      const int kn = k0 + TK;
      if constexpr (A_F32) {  // register prefetch, convert after compute
        const float4* s0p = (const float4*)(Xf + (size_t)(bm + r0) * K + kn + c0);
        const float4* s1p = (const float4*)(Xf + (size_t)(bm + r1) * K + kn + c0);
        p00 = s0p[0]; p01 = s0p[1];
        p10 = s1p[0]; p11 = s1p[1];
      } else {  // fire-and-forget async DMA into the other buffer
        copy16(Xh + (size_t)(bm + r0) * K + kn + c0, an + r0 * LDT + c0);
        copy16(Xh + (size_t)(bm + r1) * K + kn + c0, an + r1 * LDT + c0);
      }
      copy16(W + (size_t)(bn + r0) * K + kn + c0, bnx + r0 * LDT + c0);
      copy16(W + (size_t)(bn + r1) * K + kn + c0, bnx + r1 * LDT + c0);
    }
    v16bf bb0 = frag_b(Bs[cur], LDT, wn, 0);
    v16bf bb1 = frag_b(Bs[cur], LDT, wn + 16, 0);
#pragma unroll
    for (int i = 0; i < 4; ++i) {
      v16bf a = frag_a(As[cur], LDT, wm + 16 * i, 0);
      acc[i][0] = WMMA_BF16(a, bb0, acc[i][0]);
      acc[i][1] = WMMA_BF16(a, bb1, acc[i][1]);
    }
    if (nxt) {
      if constexpr (A_F32) {
        *(v8bf*)(an + r0 * LDT + c0) = pack8(p00, p01);
        *(v8bf*)(an + r1 * LDT + c0) = pack8(p10, p11);
      }
    }
    cur ^= 1;
  }

  const int col = bn + wn + (lane & 15);
  const int rbase = bm + wm + 8 * (lane >> 4);
  if constexpr (OUT_F32) {
    float* Y = (float*)Yv;
#pragma unroll
    for (int i = 0; i < 4; ++i)
#pragma unroll
      for (int j = 0; j < 2; ++j) {
        float* yp = Y + (size_t)(rbase + 16 * i) * N + col + 16 * j;
#pragma unroll
        for (int r = 0; r < 8; ++r) yp[(size_t)r * N] = acc[i][j][r];
      }
  } else {
    bf16* Y = (bf16*)Yv;
#pragma unroll
    for (int i = 0; i < 4; ++i)
#pragma unroll
      for (int j = 0; j < 2; ++j) {
        bf16* yp = Y + (size_t)(rbase + 16 * i) * N + col + 16 * j;
#pragma unroll
        for (int r = 0; r < 8; ++r) yp[(size_t)r * N] = (bf16)acc[i][j][r];
      }
  }
}

// ---------------- fused feature-map MLP, in place on q/k (bf16) ---------------
__global__ __launch_bounds__(256) void la_feature_map(bf16* __restrict__ Q,
                                                      const bf16* __restrict__ W1,
                                                      const float* __restrict__ B1,
                                                      const bf16* __restrict__ W2,
                                                      const float* __restrict__ B2) {
  constexpr int LD64 = 72, LD128 = 136;
  __shared__ bf16 Xs[64 * LD64];
  __shared__ bf16 W1s[128 * LD64];
  __shared__ bf16 Hs[64 * LD128];
  __shared__ bf16 W2s[64 * LD128];
  __shared__ float b1s[128];
  __shared__ float b2s[64];
  const int bh = blockIdx.y;
  bf16* base = Q + (size_t)(bh >> 4) * Sc * Hc + (bh & 15) * HDc;
  const int s0 = blockIdx.x * 64;
  const int lane = threadIdx.x & 31, wave = threadIdx.x >> 5;

  stage_async<64, 64, LD64>(base + (size_t)s0 * Hc, Hc, Xs);
  stage_async<128, 64, LD64>(W1, 64, W1s);
  stage_async<64, 128, LD128>(W2, 128, W2s);
  if (threadIdx.x < 128) b1s[threadIdx.x] = B1[threadIdx.x];
  if (threadIdx.x < 64) b2s[threadIdx.x] = B2[threadIdx.x];
  wait_async();
  __syncthreads();

  // layer 1: H = relu(X @ W1^T + b1), M=64 N=128 K=64
  {
    const int wm = (wave & 1) * 32, wn = (wave >> 1) * 32;
    v8f acc[2][2];
#pragma unroll
    for (int i = 0; i < 2; ++i) { zero_acc(acc[i][0]); zero_acc(acc[i][1]); }
#pragma unroll
    for (int k0 = 0; k0 < 64; k0 += 32) {
      v16bf b0 = frag_b(W1s, LD64, wn, k0);
      v16bf b1 = frag_b(W1s, LD64, wn + 16, k0);
#pragma unroll
      for (int i = 0; i < 2; ++i) {
        v16bf a = frag_a(Xs, LD64, wm + 16 * i, k0);
        acc[i][0] = WMMA_BF16(a, b0, acc[i][0]);
        acc[i][1] = WMMA_BF16(a, b1, acc[i][1]);
      }
    }
#pragma unroll
    for (int i = 0; i < 2; ++i)
#pragma unroll
      for (int j = 0; j < 2; ++j) {
        const int c = wn + 16 * j + (lane & 15);
        const float bias = b1s[c];
        const int rb = wm + 16 * i + 8 * (lane >> 4);
#pragma unroll
        for (int r = 0; r < 8; ++r) {
          float v = acc[i][j][r] + bias;
          Hs[(rb + r) * LD128 + c] = (bf16)(v > 0.0f ? v : 0.0f);
        }
      }
  }
  __syncthreads();

  // layer 2: Y = H @ W2^T + b2, M=64 N=64 K=128; write in place (bf16)
  {
    const int wm = (wave & 3) * 16, wn = (wave >> 2) * 32;
    v8f acc[2];
    zero_acc(acc[0]); zero_acc(acc[1]);
#pragma unroll
    for (int k0 = 0; k0 < 128; k0 += 32) {
      v16bf a = frag_a(Hs, LD128, wm, k0);
      acc[0] = WMMA_BF16(a, frag_b(W2s, LD128, wn, k0), acc[0]);
      acc[1] = WMMA_BF16(a, frag_b(W2s, LD128, wn + 16, k0), acc[1]);
    }
    const int rb = wm + 8 * (lane >> 4);
#pragma unroll
    for (int j = 0; j < 2; ++j) {
      const int c = wn + 16 * j + (lane & 15);
      const float bias = b2s[c];
      bf16* yp = base + (size_t)(s0 + rb) * Hc + c;
#pragma unroll
      for (int r = 0; r < 8; ++r) yp[(size_t)r * Hc] = (bf16)(acc[j][r] + bias);
    }
  }
}

// -------- kv[bh] += K^T V over an S-chunk; ksum += sum_s k (split-S) ----------
__global__ __launch_bounds__(256) void la_kv(const bf16* __restrict__ Kf,
                                             const bf16* __restrict__ V,
                                             float* __restrict__ KV,
                                             float* __restrict__ Ksum) {
  constexpr int LD64 = 72;
  __shared__ bf16 Kt[64 * LD64];  // [d][s]
  __shared__ bf16 Vt[64 * LD64];  // [e][s]
  const int bh = blockIdx.y;
  const size_t boff = (size_t)(bh >> 4) * Sc * Hc + (bh & 15) * HDc;
  const bf16* kb = Kf + boff;
  const bf16* vb = V + boff;
  const int schunk = blockIdx.x * 512;
  const int lane = threadIdx.x & 31, wave = threadIdx.x >> 5;
  const int wm = (wave & 3) * 16, wn = (wave >> 2) * 32;
  v8f acc[2];
  zero_acc(acc[0]); zero_acc(acc[1]);
  float ksacc = 0.0f;
  for (int st = 0; st < 512; st += 64) {
    __syncthreads();
    stage_T64_h(kb + (size_t)(schunk + st) * Hc, Hc, Kt, LD64);
    stage_T64_h(vb + (size_t)(schunk + st) * Hc, Hc, Vt, LD64);
    __syncthreads();
    if (threadIdx.x < 64) {
      const bf16* row = Kt + threadIdx.x * LD64;
#pragma unroll
      for (int s = 0; s < 64; ++s) ksacc += (float)row[s];
    }
#pragma unroll
    for (int k0 = 0; k0 < 64; k0 += 32) {
      v16bf a = frag_a(Kt, LD64, wm, k0);
      acc[0] = WMMA_BF16(a, frag_b(Vt, LD64, wn, k0), acc[0]);
      acc[1] = WMMA_BF16(a, frag_b(Vt, LD64, wn + 16, k0), acc[1]);
    }
  }
  float* kvb = KV + (size_t)bh * 64 * 64;
  const int col = wn + (lane & 15);
  const int rb = wm + 8 * (lane >> 4);
#pragma unroll
  for (int j = 0; j < 2; ++j)
#pragma unroll
    for (int r = 0; r < 8; ++r)
      atomicAdd(kvb + (rb + r) * 64 + col + 16 * j, acc[j][r]);
  if (threadIdx.x < 64) atomicAdd(Ksum + bh * 64 + threadIdx.x, ksacc);
}

// ---------------- attn = (q @ kv) / (q . ksum + eps) --------------------------
__global__ __launch_bounds__(256) void la_num(const bf16* __restrict__ Qf,
                                              const float* __restrict__ KV,
                                              const float* __restrict__ Ksum,
                                              bf16* __restrict__ Attn) {
  constexpr int LD64 = 72;
  __shared__ bf16 Xs[128 * LD64];
  __shared__ bf16 kvT[64 * LD64];  // [e][d]
  __shared__ float ksums[64];
  __shared__ float dens[128];
  const int bh = blockIdx.y;
  const size_t boff = (size_t)(bh >> 4) * Sc * Hc + (bh & 15) * HDc;
  const bf16* qb = Qf + boff;
  bf16* ab = Attn + boff;
  const int s0 = blockIdx.x * 128;
  stage_async<128, 64, LD64>(qb + (size_t)s0 * Hc, Hc, Xs);
  stage_T64_f32(KV + (size_t)bh * 64 * 64, 64, kvT, LD64);
  if (threadIdx.x < 64) ksums[threadIdx.x] = Ksum[bh * 64 + threadIdx.x];
  wait_async();
  __syncthreads();
  if (threadIdx.x < 128) {
    const bf16* row = Xs + threadIdx.x * LD64;
    float s = 0.0f;
#pragma unroll
    for (int d = 0; d < 64; ++d) s += (float)row[d] * ksums[d];
    dens[threadIdx.x] = 1.0f / (s + EPSc);
  }
  __syncthreads();
  const int lane = threadIdx.x & 31, wave = threadIdx.x >> 5;
  const int wm = (wave & 3) * 32, wn = (wave >> 2) * 32;
  v8f acc[2][2];
#pragma unroll
  for (int i = 0; i < 2; ++i) { zero_acc(acc[i][0]); zero_acc(acc[i][1]); }
#pragma unroll
  for (int k0 = 0; k0 < 64; k0 += 32) {
    v16bf b0 = frag_b(kvT, LD64, wn, k0);
    v16bf b1 = frag_b(kvT, LD64, wn + 16, k0);
#pragma unroll
    for (int i = 0; i < 2; ++i) {
      v16bf a = frag_a(Xs, LD64, wm + 16 * i, k0);
      acc[i][0] = WMMA_BF16(a, b0, acc[i][0]);
      acc[i][1] = WMMA_BF16(a, b1, acc[i][1]);
    }
  }
  const int col = wn + (lane & 15);
  const int rb = wm + 8 * (lane >> 4);
#pragma unroll
  for (int i = 0; i < 2; ++i)
#pragma unroll
    for (int j = 0; j < 2; ++j)
#pragma unroll
      for (int r = 0; r < 8; ++r) {
        const int row = rb + 16 * i + r;
        ab[(size_t)(s0 + row) * Hc + col + 16 * j] = (bf16)(acc[i][j][r] * dens[row]);
      }
}

// ---------------- helpers: fp32->bf16 weight conversion, zero fill ------------
__global__ __launch_bounds__(256) void la_cvt(const float* __restrict__ s,
                                              bf16* __restrict__ d, int n8) {
  const int i = blockIdx.x * 256 + threadIdx.x;
  if (i < n8) {
    const float4* p = (const float4*)(s + (size_t)i * 8);
    *(v8bf*)(d + (size_t)i * 8) = pack8(p[0], p[1]);
  }
}

__global__ void la_zero(float* __restrict__ p, int n) {
  const int i = blockIdx.x * 256 + threadIdx.x;
  if (i < n) p[i] = 0.0f;
}

extern "C" void kernel_launch(void* const* d_in, const int* in_sizes, int n_in,
                              void* d_out, int out_size, void* d_ws, size_t ws_size,
                              hipStream_t stream) {
  const float* x  = (const float*)d_in[0];
  const float* Wq = (const float*)d_in[1];
  const float* Wk = (const float*)d_in[2];
  const float* Wv = (const float*)d_in[3];
  const float* Wo = (const float*)d_in[4];
  const float* w1 = (const float*)d_in[5];
  const float* b1 = (const float*)d_in[6];
  const float* w2 = (const float*)d_in[7];
  const float* b2 = (const float*)d_in[8];

  const size_t NSZ = (size_t)Bc * Sc * Hc;   // 16,777,216
  const size_t WSZ = (size_t)Hc * Hc;        // 1,048,576
  bf16* qb  = (bf16*)d_ws;
  bf16* kb  = qb + NSZ;
  bf16* vb  = kb + NSZ;
  bf16* wqb = vb + NSZ;
  bf16* wkb = wqb + WSZ;
  bf16* wvb = wkb + WSZ;
  bf16* wob = wvb + WSZ;
  bf16* w1b = wob + WSZ;
  bf16* w2b = w1b + 2 * HDc * HDc;           // 8192 each
  float* kvf  = (float*)(w2b + 2 * HDc * HDc);
  float* ksum = kvf + 64 * 64 * 64;
  bf16* attn = vb;                           // v consumed before attn is written

  const dim3 blk(256);
  const int M = Bc * Sc;

  la_cvt<<<dim3(512), blk, 0, stream>>>(Wq, wqb, 131072);
  la_cvt<<<dim3(512), blk, 0, stream>>>(Wk, wkb, 131072);
  la_cvt<<<dim3(512), blk, 0, stream>>>(Wv, wvb, 131072);
  la_cvt<<<dim3(512), blk, 0, stream>>>(Wo, wob, 131072);
  la_cvt<<<dim3(4), blk, 0, stream>>>(w1, w1b, 1024);
  la_cvt<<<dim3(4), blk, 0, stream>>>(w2, w2b, 1024);

  const dim3 gg(Hc / 128, M / 128);  // (8, 128)
  la_gemm<true, false><<<gg, blk, 0, stream>>>(x, wqb, qb, M, Hc, Hc);
  la_gemm<true, false><<<gg, blk, 0, stream>>>(x, wkb, kb, M, Hc, Hc);
  la_gemm<true, false><<<gg, blk, 0, stream>>>(x, wvb, vb, M, Hc, Hc);

  const dim3 fg(Sc / 64, Bc * NHc);  // (64, 64)
  la_feature_map<<<fg, blk, 0, stream>>>(qb, w1b, b1, w2b, b2);
  la_feature_map<<<fg, blk, 0, stream>>>(kb, w1b, b1, w2b, b2);

  const int zn = 64 * 64 * 64 + 64 * 64;
  la_zero<<<dim3((zn + 255) / 256), blk, 0, stream>>>(kvf, zn);
  la_kv<<<dim3(8, 64), blk, 0, stream>>>(kb, vb, kvf, ksum);
  la_num<<<dim3(Sc / 128, 64), blk, 0, stream>>>(qb, kvf, ksum, attn);

  la_gemm<false, true><<<gg, blk, 0, stream>>>(attn, wob, (float*)d_out, M, Hc, Hc);
}
